// RoPEMultiheadAttention_14078902796710
// MI455X (gfx1250) — matrix operand order repaired and checked by reading, not compiled, not run on hardware
//
#include <hip/hip_runtime.h>
#include <hip/hip_bf16.h>

// ---------------------------------------------------------------------------
// RoPE multi-head attention for MI455X (gfx1250, wave32, WMMA bf16 16x16x32).
//   B=1, S=4096, D=1024, H=16, DH=64.
// Pipeline:
//   K1 gemm_bf16wmma : Xq = q Wq^T, Xk = k Wk^T, Xv = v Wv^T     (f32 out)
//   K2 rope_convert  : RoPE(Xq)*0.125 -> bf16, RoPE(Xk) -> bf16, Xv -> bf16
//   K3 flash_attn    : per (head, 128-q tile) online-softmax attention (WMMA)
//   K4 gemm_bf16wmma : out = O Wo^T + bo                          (f32 out)
// ---------------------------------------------------------------------------

typedef __attribute__((ext_vector_type(16))) __bf16 v16bf;
typedef __attribute__((ext_vector_type(8)))  float  v8f;

#define S_LEN 4096
#define D_DIM 1024
#define N_HEADS 16
#define D_HEAD 64

// ---------------------------------------------------------------------------
// K1/K4: C[M,N] = A[M,K] * B[N,K]^T (+ bias), f32 in/out, bf16 WMMA compute.
// Block: 128x128 tile, 256 threads (8 waves). Wave tile: 32x64 (2x4 frags).
// ---------------------------------------------------------------------------
__global__ __launch_bounds__(256) void gemm_bf16wmma(
    const float* __restrict__ A,   // M x K row-major
    const float* __restrict__ B,   // N x K row-major (we apply ^T)
    const float* __restrict__ bias,// N or nullptr
    float* __restrict__ C,         // M x N row-major
    int M, int N, int K)
{
    __shared__ __bf16 lA[128][34];   // +2 pad: 17-bank row stride, conflict-free
    __shared__ __bf16 lB[128][34];

    const int tid  = threadIdx.x;
    const int lane = tid & 31;
    const int w    = tid >> 5;
    const int half = lane >> 4;       // 0: lanes 0-15, 1: lanes 16-31
    const int l16  = lane & 15;

    const int m0 = blockIdx.y * 128;
    const int n0 = blockIdx.x * 128;
    const int wr = w >> 1;            // 0..3 -> row offset wr*32
    const int wc = w & 1;             // 0..1 -> col offset wc*64

    const v8f vzero = {0.f,0.f,0.f,0.f,0.f,0.f,0.f,0.f};
    v8f acc[2][4];
#pragma unroll
    for (int i = 0; i < 2; ++i)
#pragma unroll
        for (int j = 0; j < 4; ++j) acc[i][j] = vzero;

    const int ldrow = tid >> 1;        // 0..127
    const int ldcol = (tid & 1) * 16;  // 0 or 16

    for (int kt = 0; kt < K; kt += 32) {
        // ---- stage tiles (convert f32 -> bf16 on the way into LDS) ----
        const float* ap = A + (size_t)(m0 + ldrow) * K + kt + ldcol;
        const float* bp = B + (size_t)(n0 + ldrow) * K + kt + ldcol;
#pragma unroll
        for (int j = 0; j < 16; ++j) {
            lA[ldrow][ldcol + j] = (__bf16)ap[j];
            lB[ldrow][ldcol + j] = (__bf16)bp[j];
        }
        // prefetch next k-chunk (gfx1250 global_prefetch_b8)
        __builtin_prefetch(ap + 32, 0, 0);
        __builtin_prefetch(bp + 32, 0, 0);
        __syncthreads();

        // ---- A fragments: 16x32, lane holds row M=l16; ISA K striping ----
        v16bf af[2];
#pragma unroll
        for (int i = 0; i < 2; ++i) {
            const int row = wr * 32 + i * 16 + l16;
#pragma unroll
            for (int v = 0; v < 8; ++v) {
                const int kk = (half ? 8 : 0) + ((v & 4) ? 16 : 0) + (v & 3) * 2;
                af[i][2 * v]     = lA[row][kk];
                af[i][2 * v + 1] = lA[row][kk + 1];
            }
        }
        // ---- B fragments: 32x16, lane holds col N=l16; ISA K striping ----
        v16bf bfr[4];
#pragma unroll
        for (int j = 0; j < 4; ++j) {
            const int col = wc * 64 + j * 16 + l16;
#pragma unroll
            for (int v = 0; v < 8; ++v) {
                const int kk = (half ? 8 : 0) + ((v & 4) ? 16 : 0) + (v & 3) * 2;
                bfr[j][2 * v]     = lB[col][kk];
                bfr[j][2 * v + 1] = lB[col][kk + 1];
            }
        }
#pragma unroll
        for (int i = 0; i < 2; ++i)
#pragma unroll
            for (int j = 0; j < 4; ++j)
                acc[i][j] = __builtin_amdgcn_wmma_f32_16x16x32_bf16(
                    false, af[i], false, bfr[j], (short)0, acc[i][j], false, false);
        __syncthreads();
    }

    // ---- epilogue: C layout M = v + 8*half, N = l16 ----
#pragma unroll
    for (int i = 0; i < 2; ++i) {
#pragma unroll
        for (int j = 0; j < 4; ++j) {
            const int col = n0 + wc * 64 + j * 16 + l16;
            const float bv = bias ? bias[col] : 0.0f;
#pragma unroll
            for (int v = 0; v < 8; ++v) {
                const int row = m0 + wr * 32 + i * 16 + v + 8 * half;
                C[(size_t)row * N + col] = acc[i][j][v] + bv;
            }
        }
    }
}

// ---------------------------------------------------------------------------
// K2: RoPE + bf16 conversion. One thread per (even,odd) pair.
// Q additionally scaled by 1/sqrt(DH)=0.125 (folded attention scale).
// ---------------------------------------------------------------------------
__global__ __launch_bounds__(256) void rope_convert(
    const float* __restrict__ Xq, const float* __restrict__ Xk,
    const float* __restrict__ Xv,
    __bf16* __restrict__ Qb, __bf16* __restrict__ Kb, __bf16* __restrict__ Vb)
{
    const int idx = blockIdx.x * 256 + threadIdx.x;   // 0 .. S*D/2-1
    const int s   = idx >> 9;                         // / (D/2)
    const int rem = idx & 511;
    const int j   = rem & 31;                         // rotation pair in head
    const int base = s * D_DIM + (rem >> 5) * D_HEAD + 2 * j;

    const float inv = __powf(10000.0f, -(float)(2 * j) / (float)D_HEAD);
    float sn, cs;
    __sincosf((float)s * inv, &sn, &cs);

    const float qr = Xq[base], qi = Xq[base + 1];
    Qb[base]     = (__bf16)((qr * cs - qi * sn) * 0.125f);
    Qb[base + 1] = (__bf16)((qr * sn + qi * cs) * 0.125f);

    const float kr = Xk[base], ki = Xk[base + 1];
    Kb[base]     = (__bf16)(kr * cs - ki * sn);
    Kb[base + 1] = (__bf16)(kr * sn + ki * cs);

    Vb[base]     = (__bf16)Xv[base];
    Vb[base + 1] = (__bf16)Xv[base + 1];
}

// ---------------------------------------------------------------------------
// K3: flash attention. Grid: (S/128 q-tiles, H). Block: 256 threads (8 waves).
// Each wave owns 16 query rows; key/value streamed in 64-row tiles.
// ---------------------------------------------------------------------------
__global__ __launch_bounds__(256) void flash_attn(
    const __bf16* __restrict__ Qb, const __bf16* __restrict__ Kb,
    const __bf16* __restrict__ Vb, float* __restrict__ O)
{
    __shared__ __bf16 lQ[128][66];
    __shared__ __bf16 lK[64][66];
    __shared__ __bf16 lV[64][66];
    __shared__ __bf16 lP[8][16][66];   // per-wave P relayout buffer

    const int tid  = threadIdx.x;
    const int lane = tid & 31;
    const int w    = tid >> 5;
    const int half = lane >> 4;
    const int l16  = lane & 15;
    const int qt   = blockIdx.x;
    const int h    = blockIdx.y;

    // ---- load Q tile 128x64 (already bf16, RoPEd, pre-scaled) ----
    {
        const int r  = tid >> 1;
        const int c0 = (tid & 1) * 32;
        const __bf16* src = Qb + (size_t)(qt * 128 + r) * D_DIM + h * D_HEAD + c0;
#pragma unroll
        for (int j = 0; j < 32; ++j) lQ[r][c0 + j] = src[j];
    }
    __syncthreads();

    // ---- Q A-fragments held in registers for the whole kernel ----
    v16bf aq[2];
#pragma unroll
    for (int ks = 0; ks < 2; ++ks) {
        const int row = w * 16 + l16;
#pragma unroll
        for (int v = 0; v < 8; ++v) {
            const int kk = ks * 32 + (half ? 8 : 0) + ((v & 4) ? 16 : 0) + (v & 3) * 2;
            aq[ks][2 * v]     = lQ[row][kk];
            aq[ks][2 * v + 1] = lQ[row][kk + 1];
        }
    }

    const v8f vzero = {0.f,0.f,0.f,0.f,0.f,0.f,0.f,0.f};
    v8f o_acc[4];
#pragma unroll
    for (int jd = 0; jd < 4; ++jd) o_acc[jd] = vzero;
    float m[8], l[8];
#pragma unroll
    for (int v = 0; v < 8; ++v) { m[v] = -1e30f; l[v] = 0.0f; }

    for (int t = 0; t < S_LEN; t += 64) {
        __syncthreads();   // previous iteration done with lK/lV
        // ---- stage K,V tiles (64x64 bf16 each) ----
        {
            const int r  = tid >> 2;         // 0..63
            const int c0 = (tid & 3) * 16;   // 0,16,32,48
            const __bf16* ksrc = Kb + (size_t)(t + r) * D_DIM + h * D_HEAD + c0;
            const __bf16* vsrc = Vb + (size_t)(t + r) * D_DIM + h * D_HEAD + c0;
#pragma unroll
            for (int j = 0; j < 16; ++j) {
                lK[r][c0 + j] = ksrc[j];
                lV[r][c0 + j] = vsrc[j];
            }
        }
        __syncthreads();

        // ---- S = Q K^T : 16 queries x 64 keys per wave ----
        v8f sf[4];
#pragma unroll
        for (int jn = 0; jn < 4; ++jn) {
            v16bf bk[2];
#pragma unroll
            for (int ks = 0; ks < 2; ++ks) {
                const int key = jn * 16 + l16;
#pragma unroll
                for (int v = 0; v < 8; ++v) {
                    const int kk = ks * 32 + (half ? 8 : 0) + ((v & 4) ? 16 : 0) + (v & 3) * 2;
                    bk[ks][2 * v]     = lK[key][kk];
                    bk[ks][2 * v + 1] = lK[key][kk + 1];
                }
            }
            v8f z = vzero;
            z = __builtin_amdgcn_wmma_f32_16x16x32_bf16(false, aq[0], false, bk[0],
                                                        (short)0, z, false, false);
            z = __builtin_amdgcn_wmma_f32_16x16x32_bf16(false, aq[1], false, bk[1],
                                                        (short)0, z, false, false);
            sf[jn] = z;
        }

        // ---- online softmax (rows live in 16-lane halves) ----
        float mnew[8];
#pragma unroll
        for (int v = 0; v < 8; ++v) {
            float mx = m[v];
#pragma unroll
            for (int jn = 0; jn < 4; ++jn) mx = fmaxf(mx, sf[jn][v]);
#pragma unroll
            for (int d = 1; d < 16; d <<= 1) mx = fmaxf(mx, __shfl_xor(mx, d, 32));
            mnew[v] = mx;
        }
        float alpha[8];
#pragma unroll
        for (int v = 0; v < 8; ++v) {
            alpha[v] = __expf(m[v] - mnew[v]);
            m[v] = mnew[v];
            l[v] *= alpha[v];
        }
#pragma unroll
        for (int jd = 0; jd < 4; ++jd)
#pragma unroll
            for (int v = 0; v < 8; ++v) o_acc[jd][v] *= alpha[v];

        // ---- P = exp(S - m); write bf16 P to LDS for C->A relayout ----
#pragma unroll
        for (int jn = 0; jn < 4; ++jn) {
#pragma unroll
            for (int v = 0; v < 8; ++v) {
                const float p = __expf(sf[jn][v] - m[v]);
                sf[jn][v] = p;
                lP[w][v + 8 * half][jn * 16 + l16] = (__bf16)p;
            }
        }
#pragma unroll
        for (int v = 0; v < 8; ++v) {
            float rs = sf[0][v] + sf[1][v] + sf[2][v] + sf[3][v];
#pragma unroll
            for (int d = 1; d < 16; d <<= 1) rs += __shfl_xor(rs, d, 32);
            l[v] += rs;
        }
        __syncthreads();

        // ---- O += P V ----
        v16bf ap[2];
#pragma unroll
        for (int ks = 0; ks < 2; ++ks) {
#pragma unroll
            for (int v = 0; v < 8; ++v) {
                const int kk = ks * 32 + (half ? 8 : 0) + ((v & 4) ? 16 : 0) + (v & 3) * 2;
                ap[ks][2 * v]     = lP[w][l16][kk];
                ap[ks][2 * v + 1] = lP[w][l16][kk + 1];
            }
        }
#pragma unroll
        for (int jd = 0; jd < 4; ++jd) {
            v16bf bv[2];
#pragma unroll
            for (int ks = 0; ks < 2; ++ks) {
#pragma unroll
                for (int v = 0; v < 8; ++v) {
                    const int key = ks * 32 + (half ? 8 : 0) + ((v & 4) ? 16 : 0) + (v & 3) * 2;
                    bv[ks][2 * v]     = lV[key][jd * 16 + l16];
                    bv[ks][2 * v + 1] = lV[key + 1][jd * 16 + l16];
                }
            }
            o_acc[jd] = __builtin_amdgcn_wmma_f32_16x16x32_bf16(
                false, ap[0], false, bv[0], (short)0, o_acc[jd], false, false);
            o_acc[jd] = __builtin_amdgcn_wmma_f32_16x16x32_bf16(
                false, ap[1], false, bv[1], (short)0, o_acc[jd], false, false);
        }
    }

    // ---- normalize and store: out[s][h*64 + dh], f32 ----
#pragma unroll
    for (int jd = 0; jd < 4; ++jd) {
        const int col = h * D_HEAD + jd * 16 + l16;
#pragma unroll
        for (int v = 0; v < 8; ++v) {
            const int row = qt * 128 + w * 16 + v + 8 * half;
            O[(size_t)row * D_DIM + col] = o_acc[jd][v] / l[v];
        }
    }
}

// ---------------------------------------------------------------------------
// Host-side orchestration (graph-capture safe: only async launches on stream)
// ---------------------------------------------------------------------------
extern "C" void kernel_launch(void* const* d_in, const int* in_sizes, int n_in,
                              void* d_out, int out_size, void* d_ws, size_t ws_size,
                              hipStream_t stream) {
    (void)in_sizes; (void)n_in; (void)out_size; (void)ws_size;

    const float* q  = (const float*)d_in[0];
    const float* k  = (const float*)d_in[1];
    const float* v  = (const float*)d_in[2];
    const float* Wq = (const float*)d_in[3];
    const float* Wk = (const float*)d_in[4];
    const float* Wv = (const float*)d_in[5];
    const float* Wo = (const float*)d_in[6];
    const float* bo = (const float*)d_in[7];
    float* out = (float*)d_out;

    const size_t SD = (size_t)S_LEN * D_DIM;   // 4M elements

    // workspace layout (~88 MB)
    float*  Xq = (float*)d_ws;
    float*  Xk = Xq + SD;
    float*  Xv = Xk + SD;
    __bf16* Qb = (__bf16*)(Xv + SD);
    __bf16* Kb = Qb + SD;
    __bf16* Vb = Kb + SD;
    float*  Oa = (float*)(Vb + SD);

    const dim3 gGemm(D_DIM / 128, S_LEN / 128);   // (8, 32)
    const dim3 bBlk(256);

    // QKV projections (f32 -> f32, bf16 WMMA inside)
    gemm_bf16wmma<<<gGemm, bBlk, 0, stream>>>(q, Wq, (const float*)nullptr, Xq,
                                              S_LEN, D_DIM, D_DIM);
    gemm_bf16wmma<<<gGemm, bBlk, 0, stream>>>(k, Wk, (const float*)nullptr, Xk,
                                              S_LEN, D_DIM, D_DIM);
    gemm_bf16wmma<<<gGemm, bBlk, 0, stream>>>(v, Wv, (const float*)nullptr, Xv,
                                              S_LEN, D_DIM, D_DIM);

    // RoPE + bf16 conversion (scale folded into Q)
    const int ropeBlocks = (int)(SD / 2 / 256);   // 8192
    rope_convert<<<ropeBlocks, bBlk, 0, stream>>>(Xq, Xk, Xv, Qb, Kb, Vb);

    // attention
    const dim3 gAttn(S_LEN / 128, N_HEADS);       // (32, 16)
    flash_attn<<<gAttn, bBlk, 0, stream>>>(Qb, Kb, Vb, Oa);

    // output projection + bias
    gemm_bf16wmma<<<gGemm, bBlk, 0, stream>>>(Oa, Wo, bo, out,
                                              S_LEN, D_DIM, D_DIM);
}
